// GraphConv_64029372449311
// MI455X (gfx1250) — compile-verified, matrix operand order checked by reading, not run on hardware
//
#include <hip/hip_runtime.h>
#include <hip/hip_bf16.h>

// ---------------------------------------------------------------------------
// GraphConv on MI455X (gfx1250):
//   out = relu( G @ (F @ W^T) + b )   (reassociated: avoids second big GEMM)
//   G: [16384,16384] f32 (1 GiB, streamed from HBM exactly once)
//   Pt = (F @ W^T)^T stored bf16 [256][16384] in d_ws (8 MiB, L2-resident)
//   Big GEMM: v_wmma_f32_16x16x32_bf16, in-register f32->bf16 convert for G,
//   B tile staged in LDS via async global->LDS (double buffered, ASYNCcnt).
// ---------------------------------------------------------------------------

typedef __bf16 bf16_t;
typedef __attribute__((ext_vector_type(16))) __bf16 v16bf;
typedef __attribute__((ext_vector_type(8)))  __bf16 v8bf;
typedef __attribute__((ext_vector_type(8)))  float  v8f;
typedef __attribute__((ext_vector_type(4)))  float  v4f;

#define GN    16384   // graph dim / node count
#define FEATS 256     // in = out feature count
#define LROW  40      // padded LDS row stride (bf16 elems) -> conflict-free b128
#define LBUF  (256 * LROW)   // one K-chunk buffer: 256 cols x 32 K (20 KB w/ pad)

// CDNA5 16-bit A-operand layout (16x32, MxK) — and identically the B-operand
// through B^T row-major:
//   lane l (l<16):  row r+l,    K = k0+{0..7} and k0+{16..23}
//   lane l (l>=16): row r+l-16, K = k0+{8..15} and k0+{24..31}

__device__ __forceinline__ v16bf load_k16_f32(const float* __restrict__ X, int ld,
                                              int r, int k0, int lane) {
  const float* p = X + (size_t)(r + (lane & 15)) * (size_t)ld
                     + (size_t)(k0 + ((lane & 16) ? 8 : 0));
  v4f f0 = *(const v4f*)(p + 0);
  v4f f1 = *(const v4f*)(p + 4);
  v4f f2 = *(const v4f*)(p + 16);
  v4f f3 = *(const v4f*)(p + 20);
  v16bf a;
  a[0]  = (__bf16)f0[0]; a[1]  = (__bf16)f0[1]; a[2]  = (__bf16)f0[2]; a[3]  = (__bf16)f0[3];
  a[4]  = (__bf16)f1[0]; a[5]  = (__bf16)f1[1]; a[6]  = (__bf16)f1[2]; a[7]  = (__bf16)f1[3];
  a[8]  = (__bf16)f2[0]; a[9]  = (__bf16)f2[1]; a[10] = (__bf16)f2[2]; a[11] = (__bf16)f2[3];
  a[12] = (__bf16)f3[0]; a[13] = (__bf16)f3[1]; a[14] = (__bf16)f3[2]; a[15] = (__bf16)f3[3];
  return a;
}

__device__ __forceinline__ v8f wmma_bf16(v16bf a, v16bf b, v8f c) {
  // (neg_a, A, neg_b, B, c_mod, C, reuse_a, reuse_b)
  return __builtin_amdgcn_wmma_f32_16x16x32_bf16(false, a, false, b,
                                                 (short)0, c, false, false);
}

// ---------------------------------------------------------------------------
// Kernel 1:  Pt[n][k] = sum_i W[n][i] * F[k][i]   (C = W * F^T, K=256)
// grid (16, 16); block 256 (8 waves). Wave: 16 n-rows x 128 k-cols (8 tiles).
// ---------------------------------------------------------------------------
__global__ void __launch_bounds__(256)
pt_kernel(const float* __restrict__ W, const float* __restrict__ F,
          bf16_t* __restrict__ Pt) {
  const int lane    = threadIdx.x & 31;
  const int wave    = threadIdx.x >> 5;
  const int rowBase = blockIdx.y * 16;                 // n
  const int colBase = blockIdx.x * 1024 + wave * 128;  // k

  v8f acc[8];
#pragma unroll
  for (int t = 0; t < 8; ++t) acc[t] = (v8f){0.f,0.f,0.f,0.f,0.f,0.f,0.f,0.f};

#pragma unroll
  for (int k0 = 0; k0 < FEATS; k0 += 32) {
    v16bf a = load_k16_f32(W, FEATS, rowBase, k0, lane);
#pragma unroll
    for (int t = 0; t < 8; ++t) {
      v16bf b = load_k16_f32(F, FEATS, colBase + 16 * t, k0, lane);
      acc[t] = wmma_bf16(a, b, acc[t]);
    }
  }

  const int kc   = lane & 15;
  const int mAdd = (lane & 16) ? 8 : 0;
#pragma unroll
  for (int t = 0; t < 8; ++t) {
#pragma unroll
    for (int i = 0; i < 8; ++i) {
      const int n = rowBase + mAdd + i;
      const int k = colBase + 16 * t + kc;
      Pt[(size_t)n * GN + k] = (__bf16)acc[t][i];
    }
  }
}

// ---------------------------------------------------------------------------
// Kernel 2:  out[m][n] = relu( sum_k G[m][k] * Pt[n][k] + bias[n] )
// grid (256); block 256 (8 waves). Workgroup: 64 rows x 256 cols -> G read
// from HBM exactly once. The 32-K x 256-col Pt tile (16 KB) is staged in LDS
// with async global->LDS loads, double buffered; waves read it conflict-free
// with ds_load_b128. Wave: 16 rows x 128 cols (8 f32 acc tiles), 512 K-steps.
// ---------------------------------------------------------------------------
__global__ void __launch_bounds__(256)
gcn_kernel(const float* __restrict__ G, const bf16_t* __restrict__ Pt,
           const float* __restrict__ bias, float* __restrict__ out) {
  __shared__ __align__(16) bf16_t lbuf[2 * LBUF];   // 40 KB

  const int tid     = threadIdx.x;
  const int lane    = tid & 31;
  const int wave    = tid >> 5;
  const int rowBase = blockIdx.x * 64 + (wave >> 1) * 16;
  const int colBase = (wave & 1) * 128;

  // Staging: thread t owns Pt row n=t; copies 32 K-values (64 B) per chunk
  // as 4 async b128 gathers directly into LDS (tracked by ASYNCcnt).
  const bf16_t* ptRow    = Pt + (size_t)tid * GN;
  const unsigned ldsRow0 = (unsigned)(size_t)&lbuf[(size_t)tid * LROW];
  const unsigned ldsRow1 = (unsigned)(size_t)&lbuf[LBUF + (size_t)tid * LROW];

  auto stage = [&](int k0, unsigned dst) {
    const bf16_t* s = ptRow + k0;
#pragma unroll
    for (int j = 0; j < 4; ++j) {
      asm volatile("global_load_async_to_lds_b128 %0, %1, off"
                   :: "v"(dst + 16u * j), "v"(s + 8 * j)
                   : "memory");
    }
  };

  stage(0, ldsRow0);

  v8f acc[8];
#pragma unroll
  for (int t = 0; t < 8; ++t) acc[t] = (v8f){0.f,0.f,0.f,0.f,0.f,0.f,0.f,0.f};

  const float* gRow = G + (size_t)(rowBase + (lane & 15)) * GN;

  for (int c = 0; c < GN / 32; ++c) {
    // my async copies have landed; barrier makes the whole tile visible.
    asm volatile("s_wait_asynccnt 0x0" ::: "memory");
    __syncthreads();
    // Restage the other buffer (its readers finished in iteration c-1, and
    // the barrier above orders their ds reads before these async writes).
    if (c + 1 < GN / 32)
      stage((c + 1) * 32, ((c + 1) & 1) ? ldsRow1 : ldsRow0);

    const int k0 = c * 32;
    if (k0 + 256 < GN) __builtin_prefetch(gRow + k0 + 256, 0, 0);  // G stream-ahead

    v16bf a = load_k16_f32(G, GN, rowBase, k0, lane);

    const bf16_t* base = &lbuf[(size_t)(c & 1) * LBUF] + ((lane & 16) ? 8 : 0);
#pragma unroll
    for (int t = 0; t < 8; ++t) {
      const bf16_t* p = base + (size_t)(colBase + 16 * t + (lane & 15)) * LROW;
      v8bf lo = *(const v8bf*)(p + 0);    // ds_load_b128, conflict-free (pad 40)
      v8bf hi = *(const v8bf*)(p + 16);
      v16bf bm;
#pragma unroll
      for (int i = 0; i < 8; ++i) { bm[i] = lo[i]; bm[8 + i] = hi[i]; }
      acc[t] = wmma_bf16(a, bm, acc[t]);
    }
  }

  const int nc   = lane & 15;
  const int mAdd = (lane & 16) ? 8 : 0;
#pragma unroll
  for (int t = 0; t < 8; ++t) {
    const int col = colBase + 16 * t + nc;
    const float bv = bias[col];
#pragma unroll
    for (int i = 0; i < 8; ++i) {
      float v = acc[t][i] + bv;
      out[(size_t)(rowBase + mAdd + i) * FEATS + col] = v > 0.f ? v : 0.f;
    }
  }
}

// ---------------------------------------------------------------------------
extern "C" void kernel_launch(void* const* d_in, const int* in_sizes, int n_in,
                              void* d_out, int out_size, void* d_ws, size_t ws_size,
                              hipStream_t stream) {
  (void)in_sizes; (void)n_in; (void)out_size; (void)ws_size;
  const float* G  = (const float*)d_in[0];   // [16384,16384]
  const float* F  = (const float*)d_in[1];   // [16384,256]
  const float* W  = (const float*)d_in[2];   // [256,256]
  const float* b  = (const float*)d_in[3];   // [256]
  float*       out = (float*)d_out;          // [16384,256]
  bf16_t*      Pt  = (bf16_t*)d_ws;          // [256][16384] bf16 = 8 MiB

  pt_kernel<<<dim3(16, 16), 256, 0, stream>>>(W, F, Pt);
  gcn_kernel<<<dim3(256), 256, 0, stream>>>(G, Pt, b, out);
}
